// DCT2_24816321037010
// MI455X (gfx1250) — matrix operand order repaired and checked by reading, not compiled, not run on hardware
//
#include <hip/hip_runtime.h>

// 2D DCT-II (4096x4096, fp32) as two dense GEMMs against the DCT basis.
// fp32 emulated via split-bf16 (hi+lo) on V_WMMA_F32_16X16X32_BF16:
//     a*b ~= aH*bH + aH*bL + aL*bH     (f32 accumulation, ~2^-16 product err)
// CDNA5 data path: double-buffered LDS fed by GLOBAL_LOAD_ASYNC_TO_LDS_B128
// (ASYNCcnt) so HBM->LDS copies overlap the XDL WMMA pipe; only GEMM1's f32
// A operand (which needs an on-the-fly hi/lo split) uses a software-pipelined
// register staging path instead.
//
//   D[k][m] = cos(pi*k*(2m+1)/8192),  D = Dhi + Dlo (bf16 pair, built once)
//   Y = X * D^T   -> stored transposed as bf16 hi/lo  (Yt)
//   Z = D * Y     -> f32 d_out

typedef __attribute__((ext_vector_type(8)))  float  v8f;
typedef __attribute__((ext_vector_type(16))) __bf16 v16bf;
typedef __attribute__((ext_vector_type(8)))  __bf16 v8bf;
typedef __attribute__((ext_vector_type(4)))  __bf16 v4bf;

#define TN   4096          // problem size
#define BT   64            // block output tile (M and N)
#define KB   32            // K slab = one WMMA k-step
#define LSTR (KB + 8)      // LDS row stride (bf16): 40 -> conflict-free b128 bursts
#define TILE (BT * LSTR)   // elems per operand tile array

// ---------------------------------------------------------------------------
// CDNA5 async copy: per-lane 16B global -> LDS, tracked by ASYNCcnt.
// VGLOBAL encoding: VDST = LDS byte address (low 32 bits of generic LDS ptr).
// ---------------------------------------------------------------------------
static __device__ __forceinline__ void async_b128(const void* gsrc, void* ldst) {
  unsigned lds = (unsigned)(size_t)ldst;
  unsigned long long ga = (unsigned long long)(size_t)gsrc;
  asm volatile("global_load_async_to_lds_b128 %0, %1, off"
               :: "v"(lds), "v"(ga) : "memory");
}
static __device__ __forceinline__ void wait_async0() {
  asm volatile("s_wait_asynccnt 0x0" ::: "memory");
}

// ---------------------------------------------------------------------------
// Kernel 1: DCT-II basis as bf16 hi/lo. Exact integer phase reduction keeps
// the cos argument in [0, 2*pi).
// ---------------------------------------------------------------------------
__global__ __launch_bounds__(256) void dct_basis_kernel(__bf16* __restrict__ Dhi,
                                                        __bf16* __restrict__ Dlo) {
  int idx = blockIdx.x * blockDim.x + threadIdx.x;   // 0 .. 16M-1
  int k = idx >> 12;
  int m = idx & (TN - 1);
  int t = (k * (2 * m + 1)) & (4 * TN - 1);          // mod 16384
  const float step = 3.14159265358979323846f / (2.0f * (float)TN);
  float c = __cosf((float)t * step);
  __bf16 h = (__bf16)c;
  Dhi[idx] = h;
  Dlo[idx] = (__bf16)(c - (float)h);
}

// Fragment loaders (LDS staged [row][k], k-contiguous).
//   A frag: elems 0..7 = K ab..ab+7, elems 8..15 = K 16+ab..   (gap 16)
//   B frag: elems 0..7 = K kbase..kbase+7, elems 8..15 = K 16+kbase.. (gap 16)
static __device__ __forceinline__ v16bf load_frag(const __bf16* p, int gap) {
  v8bf c0 = *(const v8bf*)p;
  v8bf c1 = *(const v8bf*)(p + gap);
  return __builtin_shufflevector(c0, c1, 0, 1, 2, 3, 4, 5, 6, 7,
                                 8, 9, 10, 11, 12, 13, 14, 15);
}

// ---------------------------------------------------------------------------
// Kernel 2: C[m][n] = sum_k A[m][k] * Bsrc[n][k]   (B given pre-transposed)
//   A: Af32[m][k] split on the fly, or Ahi/Alo[m][k] pre-split (async path)
//   C: splitC ? C^T as bf16 hi/lo : f32 C
// 128 threads = 4 waves; 64x64 block tile; 32x32 per wave; double-buffered LDS.
// ---------------------------------------------------------------------------
__global__ __launch_bounds__(128) void dct_gemm_bf16_kernel(
    const float*  __restrict__ Af32,
    const __bf16* __restrict__ Ahi, const __bf16* __restrict__ Alo,
    const __bf16* __restrict__ Bhi, const __bf16* __restrict__ Blo,
    float* __restrict__ Cf32,
    __bf16* __restrict__ Chi, __bf16* __restrict__ Clo,
    int aIsF32, int splitC)
{
  // [buffer][operand: AsH,AsL,BsH,BsL][row*LSTR + k]  -> 2*4*2560*2B = 40 KB
  __shared__ __align__(16) __bf16 smem[2][4][TILE];

  const int tid   = threadIdx.x;
  const int lane  = tid & 31;
  const int wave  = tid >> 5;
  const int waveM = (wave >> 1) * 32;
  const int waveN = (wave & 1)  * 32;
  const int rowBase = blockIdx.y * BT;
  const int colBase = blockIdx.x * BT;

  const int lm    = lane & 15;
  const int ab    = (lane < 16) ? 0 : 8;    // bf16 A layout
  const int kbase = (lane < 16) ? 0 : 8;    // bf16 B layout

  // staging coordinates (b128 chunks: 2 per thread per bf16 tile array)
  const int sr0 = tid >> 2,               sk0 = (tid & 3) << 3;          // i=0
  const int sr1 = (128 + tid) >> 2,       sk1 = ((128 + tid) & 3) << 3;  // i=1

  v8f acc[2][2] = {};
  float4 a4[4];   // register pipeline for the f32-A split path

  // ---- staging helpers -----------------------------------------------------
  auto stageB_async = [&](int buf, int kk) {
    size_t g0 = (size_t)(colBase + sr0) * TN + kk + sk0;
    size_t g1 = (size_t)(colBase + sr1) * TN + kk + sk1;
    async_b128(Bhi + g0, &smem[buf][2][sr0 * LSTR + sk0]);
    async_b128(Blo + g0, &smem[buf][3][sr0 * LSTR + sk0]);
    async_b128(Bhi + g1, &smem[buf][2][sr1 * LSTR + sk1]);
    async_b128(Blo + g1, &smem[buf][3][sr1 * LSTR + sk1]);
  };
  auto stageA_async = [&](int buf, int kk) {
    size_t g0 = (size_t)(rowBase + sr0) * TN + kk + sk0;
    size_t g1 = (size_t)(rowBase + sr1) * TN + kk + sk1;
    async_b128(Ahi + g0, &smem[buf][0][sr0 * LSTR + sk0]);
    async_b128(Alo + g0, &smem[buf][0 + 1][sr0 * LSTR + sk0]);
    async_b128(Ahi + g1, &smem[buf][0][sr1 * LSTR + sk1]);
    async_b128(Alo + g1, &smem[buf][1][sr1 * LSTR + sk1]);
  };
  auto stageAf32_load = [&](int kk) {       // phase 1: global float4 -> regs
#pragma unroll
    for (int i = 0; i < 4; ++i) {
      int linear = i * 128 + tid;
      int m  = linear >> 3;
      int kq = (linear & 7) << 2;
      a4[i] = *(const float4*)(Af32 + (size_t)(rowBase + m) * TN + kk + kq);
    }
  };
  auto stageAf32_store = [&](int buf) {     // phase 2: split hi/lo -> LDS
#pragma unroll
    for (int i = 0; i < 4; ++i) {
      int linear = i * 128 + tid;
      int m  = linear >> 3;
      int kq = (linear & 7) << 2;
      float4 v = a4[i];
      __bf16 h0 = (__bf16)v.x, h1 = (__bf16)v.y, h2 = (__bf16)v.z, h3 = (__bf16)v.w;
      v4bf hv = {h0, h1, h2, h3};
      v4bf lv = {(__bf16)(v.x - (float)h0), (__bf16)(v.y - (float)h1),
                 (__bf16)(v.z - (float)h2), (__bf16)(v.w - (float)h3)};
      *(v4bf*)&smem[buf][0][m * LSTR + kq] = hv;
      *(v4bf*)&smem[buf][1][m * LSTR + kq] = lv;
    }
  };

  // ---- prologue: stage slab 0 into buffer 0 -------------------------------
  if (aIsF32) { stageAf32_load(0); stageAf32_store(0); }
  else        { stageA_async(0, 0); }
  stageB_async(0, 0);
  wait_async0();
  __syncthreads();

  int buf = 0;
  for (int kk = 0; kk < TN; kk += KB) {
    const int nxt = kk + KB;
    const bool hasNext = nxt < TN;

    // ---- issue next slab's copies before computing this one
    if (hasNext) {
      if (aIsF32) stageAf32_load(nxt);      // global loads in flight over compute
      else        stageA_async(buf ^ 1, nxt);
      stageB_async(buf ^ 1, nxt);
    }

    // ---- compute current slab: 12 WMMAs (2x2 tiles x 3 split products)
    {
      const __bf16* AH = smem[buf][0];
      const __bf16* AL = smem[buf][1];
      const __bf16* BH = smem[buf][2];
      const __bf16* BL = smem[buf][3];
      v16bf aH[2], aL[2], bH[2], bL[2];
#pragma unroll
      for (int mi = 0; mi < 2; ++mi) {
        int m = waveM + mi * 16 + lm;
        aH[mi] = load_frag(&AH[m * LSTR + ab], 16);
        aL[mi] = load_frag(&AL[m * LSTR + ab], 16);
      }
#pragma unroll
      for (int ni = 0; ni < 2; ++ni) {
        int n = waveN + ni * 16 + lm;
        bH[ni] = load_frag(&BH[n * LSTR + kbase], 16);
        bL[ni] = load_frag(&BL[n * LSTR + kbase], 16);
      }
#pragma unroll
      for (int mi = 0; mi < 2; ++mi)
#pragma unroll
        for (int ni = 0; ni < 2; ++ni) {
          v8f c = acc[mi][ni];
          c = __builtin_amdgcn_wmma_f32_16x16x32_bf16(false, aH[mi], false, bH[ni],
                                                      (short)0, c, false, false);
          c = __builtin_amdgcn_wmma_f32_16x16x32_bf16(false, aH[mi], false, bL[ni],
                                                      (short)0, c, false, false);
          c = __builtin_amdgcn_wmma_f32_16x16x32_bf16(false, aL[mi], false, bH[ni],
                                                      (short)0, c, false, false);
          acc[mi][ni] = c;
        }
    }

    // ---- finish next slab's staging, flip buffers
    if (hasNext && aIsF32) stageAf32_store(buf ^ 1);
    wait_async0();          // own async copies (into buf^1) complete
    __syncthreads();        // everyone's copies visible, everyone done reading buf
    buf ^= 1;
  }

  // ---- epilogue: C/D 16x16 f32 layout, VGPR r holds M = r + (lane<16?0:8)
  const int mh = (lane < 16) ? 0 : 8;
#pragma unroll
  for (int mi = 0; mi < 2; ++mi)
#pragma unroll
    for (int ni = 0; ni < 2; ++ni)
#pragma unroll
      for (int r = 0; r < 8; ++r) {
        int row = rowBase + waveM + mi * 16 + mh + r;
        int col = colBase + waveN + ni * 16 + lm;
        float v = acc[mi][ni][r];
        if (splitC) {                       // write C^T bf16 hi/lo (feeds GEMM2's B)
          __bf16 h = (__bf16)v;
          size_t g = (size_t)col * TN + row;
          Chi[g] = h;
          Clo[g] = (__bf16)(v - (float)h);
        } else {
          Cf32[(size_t)row * TN + col] = v;
        }
      }
}

// ---------------------------------------------------------------------------
extern "C" void kernel_launch(void* const* d_in, const int* in_sizes, int n_in,
                              void* d_out, int out_size, void* d_ws, size_t ws_size,
                              hipStream_t stream) {
  (void)in_sizes; (void)n_in; (void)out_size; (void)ws_size;

  const float* x   = (const float*)d_in[0];
  float*       out = (float*)d_out;

  const size_t NN = (size_t)TN * TN;
  __bf16* Dhi = (__bf16*)d_ws;      // 32 MB
  __bf16* Dlo = Dhi + NN;           // 32 MB
  __bf16* Yhi = Dlo + NN;           // 32 MB (Y transposed, hi)
  __bf16* Ylo = Yhi + NN;           // 32 MB (Y transposed, lo)

  dct_basis_kernel<<<dim3((unsigned)(NN / 256)), dim3(256), 0, stream>>>(Dhi, Dlo);

  dim3 grid(TN / BT, TN / BT);
  // Y = X * D^T : A = X f32 (pipelined split), B rows = D rows; Y^T bf16 hi/lo out
  dct_gemm_bf16_kernel<<<grid, dim3(128), 0, stream>>>(
      x, nullptr, nullptr, Dhi, Dlo, nullptr, Yhi, Ylo, /*aIsF32=*/1, /*splitC=*/1);
  // Z = D * Y : A = D pre-split (async), B rows = Y^T rows; f32 out
  dct_gemm_bf16_kernel<<<grid, dim3(128), 0, stream>>>(
      nullptr, Dhi, Dlo, Yhi, Ylo, out, nullptr, nullptr, /*aIsF32=*/0, /*splitC=*/0);
}
